// SelfAttentionTF_62895501083076
// MI455X (gfx1250) — compile-verified, matrix operand order checked by reading, not run on hardware
//
#include <hip/hip_runtime.h>
#include <hip/hip_bf16.h>

typedef __bf16 bf16_t;
typedef __attribute__((ext_vector_type(8)))  __bf16 v8bf;
typedef __attribute__((ext_vector_type(16))) __bf16 v16bf;
typedef __attribute__((ext_vector_type(8)))  float  v8f;
typedef __attribute__((ext_vector_type(4)))  int    v4i;

#define EMB   128
#define HEADS 16
#define BATCH 8
#define SEQ   1024
#define NH    (EMB * HEADS)     // 2048
#define MROWS (BATCH * SEQ)     // 8192
#define QK_SCALE 0.2973017787506803f   // 128^-0.25

// Async global->LDS copies (CDNA5, tracked by ASYNCcnt). Guarded so the file
// still compiles (plain-load fallback) if this toolchain lacks the builtins.
#if defined(__HIP_DEVICE_COMPILE__) &&                                        \
    __has_builtin(__builtin_amdgcn_global_load_async_to_lds_b128) &&          \
    __has_builtin(__builtin_amdgcn_s_wait_asynccnt)
#define HAVE_ASYNC 1
#else
#define HAVE_ASYNC 0
#endif

#if HAVE_ASYNC
__device__ __forceinline__ void async_cp16(void* lds, const void* g) {
  // signature (from clang diagnostic): (v4i AS1* src, v4i AS3* dst, imm, imm)
  void* gnc = const_cast<void*>(g);
  __builtin_amdgcn_global_load_async_to_lds_b128(
      (__attribute__((address_space(1))) v4i*)gnc,
      (__attribute__((address_space(3))) v4i*)lds, 0, 0);
}
#endif

// ---------------------------------------------------------------------------
// WMMA fragment helpers (wave32, 16x16x32 bf16, per CDNA5 ISA 7.12.2)
// A 16x32 (and B supplied as B^T 16x32): lane L holds row = L%16,
// elements 0..7  = K (L/16)*8 .. +7,  elements 8..15 = K 16+(L/16)*8 .. +7.
// C/D f32 16x16: lane L holds col n = L%16; vgpr r holds row m = r + 8*(L/16).
// ---------------------------------------------------------------------------
__device__ __forceinline__ v16bf load_frag(const bf16_t* base, int ld) {
  const int lane = threadIdx.x & 31;
  const int row  = lane & 15;
  const int kh   = (lane >> 4) << 3;
  const v8bf lo = *(const v8bf*)(base + row * ld + kh);
  const v8bf hi = *(const v8bf*)(base + row * ld + 16 + kh);
  return __builtin_shufflevector(lo, hi, 0,1,2,3,4,5,6,7,8,9,10,11,12,13,14,15);
}

__device__ __forceinline__ v8f wmma_bf16(v16bf a, v16bf b, v8f c) {
  return __builtin_amdgcn_wmma_f32_16x16x32_bf16(false, a, false, b, (short)0, c,
                                                 false, false);
}

// ---------------------------------------------------------------------------
// Kernel 1: QKV projection.  C[8192,2048] = x[8192,128] @ W[128,2048]
// blockDim = 128 (4 waves); block tile 64x64; wave tile 16x64; K-step 32.
// z = 0:Wq->q (scaled), 1:Wk->k (scaled), 2:Wv->v.
// Output layout: [b, h, t, e] bf16 for contiguous attention tiles.
// (f32 -> bf16 conversion happens on the LDS staging path, so tiles here
//  cannot use async copies.)
// ---------------------------------------------------------------------------
__global__ void qkv_gemm(const float* __restrict__ x,
                         const float* __restrict__ Wq,
                         const float* __restrict__ Wk,
                         const float* __restrict__ Wv,
                         bf16_t* __restrict__ qb,
                         bf16_t* __restrict__ kb,
                         bf16_t* __restrict__ vb) {
  __shared__ bf16_t sA[64 * 32];
  __shared__ bf16_t sBT[64 * 32];

  const int z   = blockIdx.z;
  const float* W = (z == 0) ? Wq : (z == 1) ? Wk : Wv;
  bf16_t*     dst = (z == 0) ? qb : (z == 1) ? kb : vb;
  const float scale = (z < 2) ? QK_SCALE : 1.0f;

  const int tid  = threadIdx.x;
  const int wave = tid >> 5;
  const int lane = tid & 31;
  const int nlan = lane & 15;
  const int hi   = lane >> 4;
  const int row0 = blockIdx.y * 64;
  const int n0   = blockIdx.x * 64;

  v8f acc[4] = {v8f{}, v8f{}, v8f{}, v8f{}};

  for (int kk = 0; kk < 4; ++kk) {
    const int k0 = kk * 32;
    // A tile: 64 rows x 32 cols of x (f32 -> bf16)
#pragma unroll
    for (int c = 0; c < 4; ++c) {
      int u = tid + c * 128;            // 512 float4 units
      int r = u >> 3, c4 = u & 7;       // 8 float4 per 32-col row
      float4 f = *(const float4*)&x[(row0 + r) * EMB + k0 + c4 * 4];
      bf16_t* p = &sA[r * 32 + c4 * 4];
      p[0] = (bf16_t)f.x; p[1] = (bf16_t)f.y; p[2] = (bf16_t)f.z; p[3] = (bf16_t)f.w;
    }
    // B tile: 32 k-rows x 64 n-cols of W, stored transposed (B^T) in LDS
#pragma unroll
    for (int c = 0; c < 4; ++c) {
      int u = tid + c * 128;
      int r = u >> 4, c4 = u & 15;      // 16 float4 per 64-col row
      float4 f = *(const float4*)&W[(k0 + r) * NH + n0 + c4 * 4];
      int n = c4 * 4;
      sBT[(n + 0) * 32 + r] = (bf16_t)f.x;
      sBT[(n + 1) * 32 + r] = (bf16_t)f.y;
      sBT[(n + 2) * 32 + r] = (bf16_t)f.z;
      sBT[(n + 3) * 32 + r] = (bf16_t)f.w;
    }
    __syncthreads();

    v16bf a = load_frag(&sA[(wave * 16) * 32], 32);
#pragma unroll
    for (int j = 0; j < 4; ++j) {
      v16bf b = load_frag(&sBT[(j * 16) * 32], 32);
      acc[j] = wmma_bf16(a, b, acc[j]);
    }
    __syncthreads();
  }

  // scatter into [b,h,t,e] bf16
#pragma unroll
  for (int j = 0; j < 4; ++j) {
#pragma unroll
    for (int r = 0; r < 8; ++r) {
      int mrow = row0 + wave * 16 + r + 8 * hi;
      int col  = n0 + j * 16 + nlan;
      int bb = mrow >> 10, t = mrow & 1023;
      int h  = col >> 7,   e = col & 127;
      dst[(((bb * HEADS + h) * SEQ) + t) * EMB + e] = (bf16_t)(acc[j][r] * scale);
    }
  }
}

// ---------------------------------------------------------------------------
// Kernel 2: fused flash attention per (b,h).
// block = 128 threads (4 waves); each wave owns a 16-query tile; block shares
// 32-wide K/V tiles in LDS; online softmax; P bounced via per-wave LDS tile.
// K tile is a pure bf16 copy -> async global->LDS path; V tile is transposed
// in flight so it keeps the VGPR bounce.
// ---------------------------------------------------------------------------
__global__ void attn_kernel(const bf16_t* __restrict__ qg,
                            const bf16_t* __restrict__ kg,
                            const bf16_t* __restrict__ vg,
                            bf16_t* __restrict__ attn) {
  __shared__ bf16_t sK[32 * 128];      // [s][e]
  __shared__ bf16_t sVT[128 * 32];     // [e][s]  (transposed)
  __shared__ bf16_t sP[4][16 * 32];    // per-wave P tile [m][s]

  const int tid  = threadIdx.x;
  const int wave = tid >> 5;
  const int lane = tid & 31;
  const int nlan = lane & 15;
  const int hi   = lane >> 4;

  const int bh    = blockIdx.y;                 // b*16 + h
  const int b     = bh >> 4, h = bh & 15;
  const long base = (long)bh * SEQ * EMB;       // q/k/v [b,h,t,e]
  const int qrow  = blockIdx.x * 64 + wave * 16;

  // Query fragments held in registers for the whole pass (16 x 128, 4 K-chunks)
  v16bf qf[4];
#pragma unroll
  for (int kc = 0; kc < 4; ++kc)
    qf[kc] = load_frag(qg + base + (long)qrow * EMB + kc * 32, EMB);

  v8f o[8];
#pragma unroll
  for (int ec = 0; ec < 8; ++ec) o[ec] = v8f{};
  float mrow[8], lrow[8];
#pragma unroll
  for (int r = 0; r < 8; ++r) { mrow[r] = -1e30f; lrow[r] = 0.0f; }

  for (int st = 0; st < SEQ; st += 32) {
    // prefetch next tile's cachelines while this one is being staged/used
    if (st + 32 < SEQ) {
      __builtin_prefetch(&kg[base + (long)(st + 32 + lane) * EMB], 0, 0);
      __builtin_prefetch(&vg[base + (long)(st + 32 + lane) * EMB], 0, 0);
    }

    // cooperative K tile load (32x128 bf16, 16B chunks)
#if HAVE_ASYNC
#pragma unroll
    for (int c = 0; c < 4; ++c) {
      int u = tid + c * 128;            // 512 16-byte units
      int r = u >> 4, cu = u & 15;
      async_cp16(&sK[r * 128 + cu * 8],
                 &kg[base + (long)(st + r) * EMB + cu * 8]);
    }
#else
#pragma unroll
    for (int c = 0; c < 4; ++c) {
      int u = tid + c * 128;            // 512 16-byte units
      int r = u >> 4, cu = u & 15;
      *(uint4*)&sK[r * 128 + cu * 8] =
          *(const uint4*)&kg[base + (long)(st + r) * EMB + cu * 8];
    }
#endif
    // cooperative V tile load, transposed into sVT[e][s]
#pragma unroll
    for (int c = 0; c < 4; ++c) {
      int u = tid + c * 128;
      int r = u >> 4, cu = u & 15;
      uint4 d = *(const uint4*)&vg[base + (long)(st + r) * EMB + cu * 8];
      const bf16_t* pv = (const bf16_t*)&d;
#pragma unroll
      for (int j = 0; j < 8; ++j) sVT[(cu * 8 + j) * 32 + r] = pv[j];
    }
#if HAVE_ASYNC
    __builtin_amdgcn_s_wait_asynccnt(0);
#endif
    __syncthreads();

    // S = q @ k^T : two 16x16 output halves over s, K=128 in 4 chunks
    v8f s0 = v8f{}, s1 = v8f{};
#pragma unroll
    for (int kc = 0; kc < 4; ++kc) {
      v16bf b0 = load_frag(&sK[0 * 128 + kc * 32], 128);
      v16bf b1 = load_frag(&sK[16 * 128 + kc * 32], 128);
      s0 = wmma_bf16(qf[kc], b0, s0);
      s1 = wmma_bf16(qf[kc], b1, s1);
    }

    // online softmax (row reductions across the 16-lane half)
    float alpha[8];
#pragma unroll
    for (int r = 0; r < 8; ++r) {
      float cmax = fmaxf(s0[r], s1[r]);
#pragma unroll
      for (int w = 1; w < 16; w <<= 1) cmax = fmaxf(cmax, __shfl_xor(cmax, w, 32));
      float mnew = fmaxf(mrow[r], cmax);
      alpha[r] = __expf(mrow[r] - mnew);
      float p0 = __expf(s0[r] - mnew);
      float p1 = __expf(s1[r] - mnew);
      float ps = p0 + p1;
#pragma unroll
      for (int w = 1; w < 16; w <<= 1) ps += __shfl_xor(ps, w, 32);
      lrow[r] = lrow[r] * alpha[r] + ps;
      mrow[r] = mnew;
      // stash P (D-layout f32 -> A-layout bf16 via per-wave LDS tile)
      int m = r + 8 * hi;
      sP[wave][m * 32 + nlan]      = (bf16_t)p0;
      sP[wave][m * 32 + 16 + nlan] = (bf16_t)p1;
    }

    // rescale running output, then accumulate P @ V
#pragma unroll
    for (int ec = 0; ec < 8; ++ec)
#pragma unroll
      for (int r = 0; r < 8; ++r) o[ec][r] *= alpha[r];

    v16bf pf = load_frag(sP[wave], 32);
#pragma unroll
    for (int ec = 0; ec < 8; ++ec) {
      v16bf vf = load_frag(&sVT[(ec * 16) * 32], 32);
      o[ec] = wmma_bf16(pf, vf, o[ec]);
    }
    __syncthreads();   // before next K/V tile overwrite
  }

  // normalize and store to attn buffer [b, t, h*128 + e] (bf16)
  float rinv[8];
#pragma unroll
  for (int r = 0; r < 8; ++r) rinv[r] = 1.0f / lrow[r];
#pragma unroll
  for (int ec = 0; ec < 8; ++ec)
#pragma unroll
    for (int r = 0; r < 8; ++r) {
      int t = qrow + r + 8 * hi;
      int e = ec * 16 + nlan;
      attn[((long)(b * SEQ + t)) * NH + h * EMB + e] = (bf16_t)(o[ec][r] * rinv[r]);
    }
}

// ---------------------------------------------------------------------------
// Kernel 3: output projection. out[8192,128] = attn[8192,2048] @ Wu[2048,128] + bu
// A tile is already bf16 -> async copy path; B tile converts f32->bf16.
// ---------------------------------------------------------------------------
__global__ void out_gemm(const bf16_t* __restrict__ attn,
                         const float* __restrict__ Wu,
                         const float* __restrict__ bu,
                         float* __restrict__ out) {
  __shared__ bf16_t sA[64 * 32];
  __shared__ bf16_t sBT[64 * 32];

  const int tid  = threadIdx.x;
  const int wave = tid >> 5;
  const int lane = tid & 31;
  const int nlan = lane & 15;
  const int hi   = lane >> 4;
  const int row0 = blockIdx.y * 64;
  const int n0   = blockIdx.x * 64;

  v8f acc[4] = {v8f{}, v8f{}, v8f{}, v8f{}};

  for (int kk = 0; kk < 64; ++kk) {
    const int k0 = kk * 32;
    // A tile (already bf16): 64 x 32 — pure copy, use async global->LDS
#if HAVE_ASYNC
#pragma unroll
    for (int c = 0; c < 2; ++c) {
      int u = tid + c * 128;            // 256 16-byte units
      int r = u >> 2, cu = u & 3;       // 4 units per 32-col row
      async_cp16(&sA[r * 32 + cu * 8],
                 &attn[(long)(row0 + r) * NH + k0 + cu * 8]);
    }
#else
#pragma unroll
    for (int c = 0; c < 2; ++c) {
      int u = tid + c * 128;            // 256 16-byte units
      int r = u >> 2, cu = u & 3;       // 4 units per 32-col row
      *(uint4*)&sA[r * 32 + cu * 8] =
          *(const uint4*)&attn[(long)(row0 + r) * NH + k0 + cu * 8];
    }
#endif
    // B tile: Wu 32 k-rows x 64 n-cols f32, transposed into LDS as bf16
#pragma unroll
    for (int c = 0; c < 4; ++c) {
      int u = tid + c * 128;
      int r = u >> 4, c4 = u & 15;
      float4 f = *(const float4*)&Wu[(k0 + r) * EMB + n0 + c4 * 4];
      int n = c4 * 4;
      sBT[(n + 0) * 32 + r] = (bf16_t)f.x;
      sBT[(n + 1) * 32 + r] = (bf16_t)f.y;
      sBT[(n + 2) * 32 + r] = (bf16_t)f.z;
      sBT[(n + 3) * 32 + r] = (bf16_t)f.w;
    }
#if HAVE_ASYNC
    __builtin_amdgcn_s_wait_asynccnt(0);
#endif
    __syncthreads();

    v16bf a = load_frag(&sA[(wave * 16) * 32], 32);
#pragma unroll
    for (int j = 0; j < 4; ++j) {
      v16bf b = load_frag(&sBT[(j * 16) * 32], 32);
      acc[j] = wmma_bf16(a, b, acc[j]);
    }
    __syncthreads();
  }

#pragma unroll
  for (int j = 0; j < 4; ++j) {
#pragma unroll
    for (int r = 0; r < 8; ++r) {
      int mrow = row0 + wave * 16 + r + 8 * hi;
      int col  = n0 + j * 16 + nlan;
      out[(long)mrow * EMB + col] = acc[j][r] + bu[col];
    }
  }
}

// ---------------------------------------------------------------------------
extern "C" void kernel_launch(void* const* d_in, const int* in_sizes, int n_in,
                              void* d_out, int out_size, void* d_ws, size_t ws_size,
                              hipStream_t stream) {
  // setup_inputs order: x, Wk, Wq, Wv, Wu, bu
  const float* x  = (const float*)d_in[0];
  const float* Wk = (const float*)d_in[1];
  const float* Wq = (const float*)d_in[2];
  const float* Wv = (const float*)d_in[3];
  const float* Wu = (const float*)d_in[4];
  const float* bu = (const float*)d_in[5];
  float* out = (float*)d_out;

  const size_t NBHTE = (size_t)BATCH * HEADS * SEQ * EMB;   // 16Mi elems
  bf16_t* qb = (bf16_t*)d_ws;
  bf16_t* kb = qb + NBHTE;
  bf16_t* vb = kb + NBHTE;
  bf16_t* ab = vb + NBHTE;   // attention out, [b,t,h*e]

  qkv_gemm<<<dim3(NH / 64, MROWS / 64, 3), 128, 0, stream>>>(x, Wq, Wk, Wv,
                                                             qb, kb, vb);
  attn_kernel<<<dim3(SEQ / 64, BATCH * HEADS), 128, 0, stream>>>(qb, kb, vb, ab);
  out_gemm<<<dim3(EMB / 64, MROWS / 64), 128, 0, stream>>>(ab, Wu, bu, out);
}
